// ResGCN15_58128087384887
// MI455X (gfx1250) — matrix operand order repaired and checked by reading, not compiled, not run on hardware
//
#include <hip/hip_runtime.h>
#include <hip/hip_bf16.h>
#include <math.h>

// ---------------- types ----------------
typedef __attribute__((ext_vector_type(16))) __bf16 v16bf;
typedef __attribute__((ext_vector_type(4)))  __bf16 v4bf;
typedef __attribute__((ext_vector_type(8)))  float  v8f;
typedef __attribute__((ext_vector_type(4)))  float  v4f;

#define GN     50000
#define GE     800000
#define NFEAT  256
#define NHID   128
#define NCLASS 40
#define NMID   13
#define CATW   1792   // 14 * NHID

// ---------------- f32 -> bf16 weight conversion ----------------
__global__ void cvt_bf16(const float* __restrict__ s, __bf16* __restrict__ d, int n) {
    int i = blockIdx.x * blockDim.x + threadIdx.x;
    if (i < n) d[i] = (__bf16)s[i];
}

// ---------------- WMMA GEMM:  C[M x Nreal] = A[M x K] * B[K x Nreal] (+bias) ----
// A: fp32, row stride lda (converted to packed bf16 while staging into LDS)
// B: bf16, row-major, row stride ldb (= Nreal)
// Block: 256 threads = 8 waves; block tile M=128; wave tile 16 x (NT*16).
// Out-of-range rows/cols are CLAMPED during staging (branch-free); the epilogue
// masks those elements, so clamped data never reaches C.
// Staging is split into load-phase (global loads issued before the barrier,
// overlapping prior chunk's WMMAs) and store-phase (cvt + LDS stores after it).
template<int NT>
__global__ __launch_bounds__(256)
void gemm_wmma(const float* __restrict__ A, int lda,
               const __bf16* __restrict__ B, int ldb,
               float* __restrict__ C, int ldc,
               const float* __restrict__ bias,
               int M, int Nreal, int K)
{
    // rows padded to 18 dwords (36 bf16) to break 16-stride LDS bank conflicts;
    // 72-byte rows keep 8-byte alignment for b64 stores.
    constexpr int AWD = 18;                    // dwords per padded row
    __shared__ unsigned As[128 * AWD];         // A tile, [128 rows][32 bf16]
    __shared__ unsigned Bt[NT * 16 * AWD];     // B tile, TRANSPOSED [n][32 bf16 along k]

    const int tid   = threadIdx.x;
    const int wave  = tid >> 5;
    const int lane  = tid & 31;
    const int l15   = lane & 15;
    const int lhalf = lane >> 4;
    const int m0    = blockIdx.x * 128;

    v8f acc[NT];
    #pragma unroll
    for (int i = 0; i < NT; i++)
        #pragma unroll
        for (int j = 0; j < 8; j++) acc[i][j] = 0.0f;

    for (int k0 = 0; k0 < K; k0 += 32) {
        // ---- load phase: issue ALL global loads before the barrier ----
        v4f va[4];                               // A: 128x32 f32 = 4 x b128 per thread
        #pragma unroll
        for (int t = 0; t < 4; t++) {
            int idx = tid + t * 256;
            int r = idx >> 3, c4 = idx & 7;      // c4: which float4 of the 32-col row
            int gr = m0 + r;
            gr = gr < M ? gr : (M - 1);          // clamp: masked in epilogue
            va[t] = *(const v4f*)(A + (size_t)gr * lda + k0 + c4 * 4);
        }
        __bf16 hb[NT][2];                        // B: NT dwords (k-pairs) per thread
        #pragma unroll
        for (int t = 0; t < NT; t++) {
            int idx = tid + t * 256;
            int n = idx >> 4, kw = idx & 15;     // kw: dword (k-pair) index
            int nc = n < Nreal ? n : (Nreal - 1);
            hb[t][0] = B[(size_t)(k0 + 2 * kw)     * ldb + nc];
            hb[t][1] = B[(size_t)(k0 + 2 * kw + 1) * ldb + nc];
        }

        __syncthreads();   // prior chunk's LDS reads complete

        // ---- store phase: packed cvt + LDS stores ----
        #pragma unroll
        for (int t = 0; t < 4; t++) {
            int idx = tid + t * 256;
            int r = idx >> 3, c4 = idx & 7;
            union { v4bf b; uint2 u; } p;
            p.b = __builtin_convertvector(va[t], v4bf);   // 2x v_cvt_pk_bf16_f32
            *(uint2*)(As + r * AWD + c4 * 2) = p.u;
        }
        #pragma unroll
        for (int t = 0; t < NT; t++) {
            int idx = tid + t * 256;
            int n = idx >> 4, kw = idx & 15;
            union { __bf16 h[2]; unsigned u; } p;
            p.h[0] = hb[t][0];
            p.h[1] = hb[t][1];
            Bt[n * AWD + kw] = p.u;
        }

        __syncthreads();   // tile visible to all waves

        // A fragment (16x32 bf16, ISA 7.12.2): lane l15 = M, lhalf selects K half
        union { unsigned u[8]; v16bf v; } af;
        const int mr   = wave * 16 + l15;
        const int kadd = lhalf * 8;                // klo/2
        #pragma unroll
        for (int q = 0; q < 8; q++) {
            int w = q + kadd;
            af.u[q] = As[mr * AWD + w];
        }

        #pragma unroll
        for (int nt = 0; nt < NT; nt++) {
            // B fragment (32x16 bf16): lane l15 = N, lhalf selects K half
            union { unsigned u[8]; v16bf v; } bfr;
            const int nn = nt * 16 + l15;
            const int kb = lhalf * 8;
            #pragma unroll
            for (int q = 0; q < 8; q++)
                bfr.u[q] = Bt[nn * AWD + kb + q];

            acc[nt] = __builtin_amdgcn_wmma_f32_16x16x32_bf16(
                false, af.v, false, bfr.v, (short)0, acc[nt], false, false);
        }
    }

    // epilogue: C/D layout — lanes 0-15: M=r, lanes 16-31: M=8+r; N=l15 within tile
    const int rowb = m0 + wave * 16 + lhalf * 8;
    #pragma unroll
    for (int r = 0; r < 8; r++) {
        int gr = rowb + r;
        if (gr < M) {
            #pragma unroll
            for (int nt = 0; nt < NT; nt++) {
                int col = nt * 16 + l15;
                if (col < Nreal) {
                    float v = acc[nt][r];
                    if (bias) v += bias[col];
                    C[(size_t)gr * ldc + col] = v;
                }
            }
        }
    }
}

// ---------------- SpMM: agg[dst] += w_e * support[src], 128 feats ----------------
// one wave per edge, float4 per lane; agg (25.6MB) is L2-resident -> on-chip atomics
__global__ __launch_bounds__(256)
void spmm128(const int* __restrict__ src, const int* __restrict__ dst,
             const float* __restrict__ w, const float* __restrict__ support,
             float* __restrict__ agg, int E_)
{
    int tid = blockIdx.x * blockDim.x + threadIdx.x;
    int e = tid >> 5;
    if (e >= E_) return;
    int l = tid & 31;
    int s = src[e], d = dst[e];
    float we = w[e];
    const float4* sp = (const float4*)(support + (size_t)s * NHID);
    float4 v = sp[l];
    float* ag = agg + (size_t)d * NHID + l * 4;
    atomicAdd(ag + 0, we * v.x);
    atomicAdd(ag + 1, we * v.y);
    atomicAdd(ag + 2, we * v.z);
    atomicAdd(ag + 3, we * v.w);
}

// ---------------- SpMM for 40 classes ----------------
__global__ __launch_bounds__(256)
void spmm40(const int* __restrict__ src, const int* __restrict__ dst,
            const float* __restrict__ w, const float* __restrict__ support,
            float* __restrict__ agg, int E_)
{
    int tid = blockIdx.x * blockDim.x + threadIdx.x;
    if (tid >= E_ * NCLASS) return;
    int e = tid / NCLASS, f = tid - e * NCLASS;
    int s = src[e], d = dst[e];
    atomicAdd(agg + (size_t)d * NCLASS + f, w[e] * support[(size_t)s * NCLASS + f]);
}

// ---------------- x1 = relu(agg + b1) + z ; write into cat at col 13*128 -----
__global__ __launch_bounds__(256)
void finish1(const float* __restrict__ agg, const float* __restrict__ b,
             const float* __restrict__ z, float* __restrict__ cat)
{
    int i = blockIdx.x * blockDim.x + threadIdx.x;
    if (i >= GN * NHID) return;
    int r = i >> 7, c = i & 127;
    float v = agg[i] + b[c];
    v = v > 0.0f ? v : 0.0f;
    v += z[i];
    cat[(size_t)r * CATW + 13 * NHID + c] = v;
}

// ---------------- h_{k+1} = relu(agg + b) + h_k (both live in cat buffer) ------
__global__ __launch_bounds__(256)
void finish_mid(const float* __restrict__ agg, const float* __restrict__ b,
                float* cat, int offIn, int offOut)
{
    int i = blockIdx.x * blockDim.x + threadIdx.x;
    if (i >= GN * NHID) return;
    int r = i >> 7, c = i & 127;
    float h = cat[(size_t)r * CATW + offIn + c];
    float v = agg[i] + b[c];
    v = v > 0.0f ? v : 0.0f;
    cat[(size_t)r * CATW + offOut + c] = v + h;
}

// ---------------- out = log_softmax(agg + b15) per row ----------------
__global__ __launch_bounds__(256)
void logsoftmax40(const float* __restrict__ agg, const float* __restrict__ b,
                  float* __restrict__ out)
{
    int r = blockIdx.x * blockDim.x + threadIdx.x;
    if (r >= GN) return;
    float vals[NCLASS];
    float mx = -3.4e38f;
    #pragma unroll
    for (int c = 0; c < NCLASS; c++) {
        float v = agg[(size_t)r * NCLASS + c] + b[c];
        vals[c] = v;
        mx = v > mx ? v : mx;
    }
    float s = 0.0f;
    #pragma unroll
    for (int c = 0; c < NCLASS; c++) s += __expf(vals[c] - mx);
    float ls = __logf(s);
    #pragma unroll
    for (int c = 0; c < NCLASS; c++)
        out[(size_t)r * NCLASS + c] = vals[c] - mx - ls;
}

// ---------------- launcher ----------------
extern "C" void kernel_launch(void* const* d_in, const int* in_sizes, int n_in,
                              void* d_out, int out_size, void* d_ws, size_t ws_size,
                              hipStream_t stream)
{
    const float* x    = (const float*)d_in[0];
    const int*   esrc = (const int*)  d_in[1];
    const int*   edst = (const int*)  d_in[2];
    const float* ew   = (const float*)d_in[3];
    const float* W0   = (const float*)d_in[4];
    const float* b0   = (const float*)d_in[5];
    const float* W1   = (const float*)d_in[6];
    const float* b1   = (const float*)d_in[7];
    const float* Wmid = (const float*)d_in[8];
    const float* bmid = (const float*)d_in[9];
    const float* W15  = (const float*)d_in[10];
    const float* b15  = (const float*)d_in[11];
    float* out = (float*)d_out;

    char* ws = (char*)d_ws;
    size_t o = 0;
    auto take = [&](size_t bytes) -> char* {
        char* p = ws + o;
        o = (o + bytes + 255) & ~(size_t)255;
        return p;
    };
    float*  cat   = (float*) take((size_t)GN * CATW * 4);   // x14..x1 column slices
    float*  zbuf  = (float*) take((size_t)GN * NHID * 4);   // skip branch
    float*  sup   = (float*) take((size_t)GN * NHID * 4);   // h @ W   (also N x 40)
    float*  agg   = (float*) take((size_t)GN * NHID * 4);   // segment_sum (also N x 40)
    __bf16* W0b   = (__bf16*)take((size_t)NFEAT * NHID * 2);
    __bf16* W1b   = (__bf16*)take((size_t)NFEAT * NHID * 2);
    __bf16* Wmidb = (__bf16*)take((size_t)NMID * NHID * NHID * 2);
    __bf16* W15b  = (__bf16*)take((size_t)CATW * NCLASS * 2);

    // weights -> bf16
    cvt_bf16<<<(NFEAT * NHID + 255) / 256, 256, 0, stream>>>(W0, W0b, NFEAT * NHID);
    cvt_bf16<<<(NFEAT * NHID + 255) / 256, 256, 0, stream>>>(W1, W1b, NFEAT * NHID);
    cvt_bf16<<<(NMID * NHID * NHID + 255) / 256, 256, 0, stream>>>(Wmid, Wmidb, NMID * NHID * NHID);
    cvt_bf16<<<(CATW * NCLASS + 255) / 256, 256, 0, stream>>>(W15, W15b, CATW * NCLASS);

    const dim3 gblk((GN + 127) / 128);
    const int spmmBlocks   = (GE * 32 + 255) / 256;
    const int spmm40Blocks = (GE * NCLASS + 255) / 256;
    const int ewBlocks     = (GN * NHID + 255) / 256;

    // skip branch: z = x @ W0 + b0
    gemm_wmma<8><<<gblk, 256, 0, stream>>>(x, NFEAT, W0b, NHID, zbuf, NHID, b0,
                                           GN, NHID, NFEAT);
    // gc1: support = x @ W1
    gemm_wmma<8><<<gblk, 256, 0, stream>>>(x, NFEAT, W1b, NHID, sup, NHID, nullptr,
                                           GN, NHID, NFEAT);
    hipMemsetAsync(agg, 0, (size_t)GN * NHID * 4, stream);
    spmm128<<<spmmBlocks, 256, 0, stream>>>(esrc, edst, ew, sup, agg, GE);
    finish1<<<ewBlocks, 256, 0, stream>>>(agg, b1, zbuf, cat);

    // gc2..gc14 with residual; x_j stored at cat col (14-j)*128
    for (int k = 0; k < NMID; k++) {
        int offIn  = (13 - k) * NHID;   // x_{k+1}
        int offOut = (12 - k) * NHID;   // x_{k+2}
        gemm_wmma<8><<<gblk, 256, 0, stream>>>(cat + offIn, CATW,
                                               Wmidb + (size_t)k * NHID * NHID, NHID,
                                               sup, NHID, nullptr, GN, NHID, NHID);
        hipMemsetAsync(agg, 0, (size_t)GN * NHID * 4, stream);
        spmm128<<<spmmBlocks, 256, 0, stream>>>(esrc, edst, ew, sup, agg, GE);
        finish_mid<<<ewBlocks, 256, 0, stream>>>(agg, bmid + k * NHID, cat, offIn, offOut);
    }

    // gc15: support40 = cat @ W15 ; spmm ; +b15 ; log_softmax
    gemm_wmma<3><<<gblk, 256, 0, stream>>>(cat, CATW, W15b, NCLASS, sup, NCLASS,
                                           nullptr, GN, NCLASS, CATW);
    hipMemsetAsync(agg, 0, (size_t)GN * NCLASS * 4, stream);
    spmm40<<<spmm40Blocks, 256, 0, stream>>>(esrc, edst, ew, sup, agg, GE);
    logsoftmax40<<<(GN + 255) / 256, 256, 0, stream>>>(agg, b15, out);
}
